// TripleOrganism_71433896067265
// MI455X (gfx1250) — matrix operand order repaired and checked by reading, not compiled, not run on hardware
//
#include <hip/hip_runtime.h>

typedef __attribute__((ext_vector_type(16))) _Float16 v16h;
typedef __attribute__((ext_vector_type(8)))  _Float16 v8h;
typedef __attribute__((ext_vector_type(8)))  float    v8f;

#define NTOT 4608
#define NPO  1536
#define DDIM 256
#define HDIM 512

// ---------------------------------------------------------------- f32 -> f16
__global__ __launch_bounds__(256)
void f32_to_f16_kernel(const float* __restrict__ src, _Float16* __restrict__ dst, int n) {
    int i = blockIdx.x * blockDim.x + threadIdx.x;
    if (i < n) dst[i] = (_Float16)src[i];
}

// ------------------------------------------------- f32 -> f16 with transpose
// src: 3 x K x Ncols (row-major) -> dst: 3 x Ncols x K  (N-major)
// so each B-fragment lane later reads 16 contiguous halves (one 32B run).
__global__ __launch_bounds__(256)
void convert_transpose_kernel(const float* __restrict__ src, _Float16* __restrict__ dst,
                              int K, int Ncols) {
    int idx = blockIdx.x * blockDim.x + threadIdx.x;
    int total = 3 * K * Ncols;
    if (idx >= total) return;
    int org = idx / (K * Ncols);
    int rem = idx - org * K * Ncols;
    int k = rem / Ncols;
    int n = rem - k * Ncols;
    dst[((size_t)org * Ncols + n) * K + k] = (_Float16)src[idx];
}

// ---------------------------------------------------------------- WMMA GEMM
// out[row,col] = act( sum_k A[row,k] * W[org, k, col] + bias[org, col] )
// Wt is pre-transposed to 3 x Ncols x K. Each wave computes a 16x64 strip:
// 4 independent f32 accumulators per K-step -> pipelined WMMAs, A reused.
// All fragment loads are contiguous 16B/32B runs (global_load_b128).
__global__ __launch_bounds__(128)
void wmma_gemm_kernel(const _Float16* __restrict__ A,    // Nrows x K
                      const _Float16* __restrict__ Wt,   // 3 x Ncols x K
                      const float*    __restrict__ bias, // 3 x Ncols
                      float*    __restrict__ outF,       // optional f32 out
                      _Float16* __restrict__ outH,       // optional f16 out
                      int K, int Ncols, int applyTanh)
{
    const int lane = threadIdx.x & 31;
    const int g    = lane >> 4;      // lane group 0/1
    const int lr   = lane & 15;
    const int m0   = blockIdx.x * 16;
    const int n0   = (blockIdx.y * 4 + threadIdx.y) * 64;  // 64 cols per wave
    const int org  = m0 / NPO;

    v8f acc[4] = {v8f{}, v8f{}, v8f{}, v8f{}};
    const _Float16* arow  = A  + (size_t)(m0 + lr) * K;
    const _Float16* brow0 = Wt + ((size_t)org * Ncols + (n0 + lr)) * K;

    for (int k0 = 0; k0 < K; k0 += 32) {
        // A fragment: halves 0..15 -> K=k0+g*16+h (matches B order)
        // => two contiguous 16B loads.
        v8h alo = *(const v8h*)(arow + k0 + g * 16);
        v8h ahi = *(const v8h*)(arow + k0 + g * 16 + 8);
        v16h af = __builtin_shufflevector(alo, ahi,
                      0,1,2,3,4,5,6,7,8,9,10,11,12,13,14,15);

        // B fragments: lane group g holds K = k0 + g*16 + h at fixed column,
        // contiguous in the transposed weight => one aligned 32B load each.
        #pragma unroll
        for (int t = 0; t < 4; ++t) {
            v16h bf = *(const v16h*)(brow0 + (size_t)t * 16 * K + k0 + g * 16);
            acc[t] = __builtin_amdgcn_wmma_f32_16x16x32_f16(
                         false, af, false, bf, (short)0, acc[t], false, false);
        }
    }

    // C/D layout: M = g*8 + r, N = lane&15
    #pragma unroll
    for (int t = 0; t < 4; ++t) {
        #pragma unroll
        for (int r = 0; r < 8; ++r) {
            int row = m0 + g * 8 + r;
            int col = n0 + t * 16 + lr;
            float v = acc[t][r] + bias[org * Ncols + col];
            if (applyTanh) v = tanhf(v);
            if (outF) outF[(size_t)row * Ncols + col] = v;
            if (outH) outH[(size_t)row * Ncols + col] = (_Float16)v;
        }
    }
}

// ---------------------------------------------------------------- neighbors
// One wave per agent: O(N) scan of positions, ballot over same-org neighbors,
// accumulate Abar/Bbar (512 dims spread 16-per-lane), then
// net_influence[i] = (A_i . Bbar - B_i . Abar) / sqrt(H) / max(cnt,1).
__global__ __launch_bounds__(256)
void neighbor_kernel(const int* __restrict__ positions, const int* __restrict__ roles,
                     const float* __restrict__ Abuf, const float* __restrict__ Bbuf,
                     int* __restrict__ cntO, int* __restrict__ smassO,
                     int* __restrict__ sfiO, int* __restrict__ efiO,
                     float* __restrict__ ninfO)
{
    const int lane = threadIdx.x & 31;
    const int wave = threadIdx.x >> 5;
    const int i = blockIdx.x * 8 + wave;
    const int xi = positions[2 * i], yi = positions[2 * i + 1];
    const int orgi = i / NPO;

    int cnt = 0, smass = 0, sfi = 0, efi = 0;
    float bacc[16], aacc[16];
    #pragma unroll
    for (int t = 0; t < 16; ++t) { bacc[t] = 0.0f; aacc[t] = 0.0f; }

    for (int base = 0; base < NTOT; base += 32) {
        int j  = base + lane;
        int dx = positions[2 * j]     - xi;
        int dy = positions[2 * j + 1] - yi;
        bool nb   = (dx * dx + dy * dy <= 25) && (j != i);
        bool same = ((j / NPO) == orgi);
        int  rj   = roles[j];
        bool sn   = nb && same;
        cnt   += sn ? 1 : 0;
        smass += (sn && rj == 0) ? 1 : 0;
        sfi   += (sn && rj == 1) ? 1 : 0;
        efi   += (nb && !same && rj == 1) ? 1 : 0;

        unsigned long long m = __ballot(sn);   // wave32: low 32 bits
        while (m) {
            int jj = base + (__ffsll(m) - 1);
            m &= m - 1;
            const float* Bj = Bbuf + (size_t)jj * HDIM + lane;
            const float* Aj = Abuf + (size_t)jj * HDIM + lane;
            #pragma unroll
            for (int t = 0; t < 16; ++t) { bacc[t] += Bj[32 * t]; aacc[t] += Aj[32 * t]; }
        }
    }

    const float* Ai = Abuf + (size_t)i * HDIM + lane;
    const float* Bi = Bbuf + (size_t)i * HDIM + lane;
    float outd = 0.0f, ind = 0.0f;
    #pragma unroll
    for (int t = 0; t < 16; ++t) { outd += Ai[32 * t] * bacc[t]; ind += Bi[32 * t] * aacc[t]; }

    #pragma unroll
    for (int off = 16; off; off >>= 1) {
        outd  += __shfl_xor(outd,  off, 32);
        ind   += __shfl_xor(ind,   off, 32);
        cnt   += __shfl_xor(cnt,   off, 32);
        smass += __shfl_xor(smass, off, 32);
        sfi   += __shfl_xor(sfi,   off, 32);
        efi   += __shfl_xor(efi,   off, 32);
    }
    if (lane == 0) {
        cntO[i] = cnt; smassO[i] = smass; sfiO[i] = sfi; efiO[i] = efi;
        const float inv_sqrtH = 0.04419417382415922f; // 1/sqrt(512)
        float denom = (cnt > 0) ? (float)cnt : 1.0f;
        ninfO[i] = (cnt > 0) ? (outd - ind) * inv_sqrtH / denom : 0.0f;
    }
}

// ---------------------------------------------------------------- finalize
__global__ __launch_bounds__(256)
void finalize_kernel(const float* __restrict__ states, const float* __restrict__ energies,
                     const float* __restrict__ rgrid, const int* __restrict__ roles,
                     const int* __restrict__ positions, const float* __restrict__ SP,
                     const int* __restrict__ cnt, const int* __restrict__ smass,
                     const int* __restrict__ sfi, const int* __restrict__ efi,
                     const float* __restrict__ ninf, float* __restrict__ out)
{
    const int i = blockIdx.x;
    const int c = threadIdx.x;
    const bool has_nb = cnt[i] > 0;
    float s = states[(size_t)i * DDIM + c];
    if (has_nb) s += 0.1f * SP[(size_t)i * DDIM + c];
    out[(size_t)i * 261 + c] = s;
    if (c == 0) {
        int x = positions[2 * i], y = positions[2 * i + 1];
        float res = rgrid[y * 256 + x];
        int role = roles[i];
        float e  = energies[i];
        float sm = (float)smass[i], ef = (float)efi[i];
        float e_fi   = e + 0.03f * sm + 0.02f * res - 0.04f * ef - 0.01f;
        float e_mass = e * 0.98f + (sfi[i] > 0 ? 0.02f : 0.0f) + 0.01f * res - 0.02f * ef;
        float ne = (role == 1) ? e_fi : e_mass;
        ne = fminf(fmaxf(ne, 0.0f), 1.0f);
        bool can_fi = (ne > 0.5f) && (smass[i] >= 2) && (sfi[i] == 0) && (efi[i] == 0);
        bool loses  = (smass[i] < 1) || (ne < 0.15f) || (efi[i] > sfi[i] + 1);
        int nr = role;
        if (role == 0)      nr = can_fi ? 1 : 0;
        else if (role == 1) nr = loses ? 0 : 1;
        out[(size_t)i * 261 + 256] = ninf[i];
        out[(size_t)i * 261 + 257] = ne;
        out[(size_t)i * 261 + 258] = (nr == 0) ? 1.0f : 0.0f;
        out[(size_t)i * 261 + 259] = (nr == 1) ? 1.0f : 0.0f;
        out[(size_t)i * 261 + 260] = (nr == 2) ? 1.0f : 0.0f;
    }
}

// ---------------------------------------------------------------- launch
extern "C" void kernel_launch(void* const* d_in, const int* in_sizes, int n_in,
                              void* d_out, int out_size, void* d_ws, size_t ws_size,
                              hipStream_t stream) {
    const float* states   = (const float*)d_in[0];
    const float* energies = (const float*)d_in[1];
    const float* rgrid    = (const float*)d_in[2];
    const float* W_out    = (const float*)d_in[3];
    const float* b_out    = (const float*)d_in[4];
    const float* W_in     = (const float*)d_in[5];
    const float* b_in     = (const float*)d_in[6];
    const float* W1       = (const float*)d_in[7];
    const float* b1       = (const float*)d_in[8];
    const float* W2       = (const float*)d_in[9];
    const float* b2       = (const float*)d_in[10];
    const int*   roles    = (const int*)d_in[11];
    // d_in[12] org_ids unused: org = i / 1536 by construction
    const int*   positions = (const int*)d_in[13];
    float* out = (float*)d_out;

    char* w = (char*)d_ws;
    size_t off = 0;
    auto alloc = [&](size_t bytes) -> void* {
        void* p = w + off; off += (bytes + 255) & ~(size_t)255; return p;
    };
    _Float16* Sh    = (_Float16*)alloc((size_t)NTOT * DDIM * 2);
    _Float16* WoutT = (_Float16*)alloc((size_t)3 * DDIM * HDIM * 2);  // 3 x H x D
    _Float16* WinT  = (_Float16*)alloc((size_t)3 * DDIM * HDIM * 2);
    _Float16* W1T   = (_Float16*)alloc((size_t)3 * DDIM * HDIM * 2);
    _Float16* W2T   = (_Float16*)alloc((size_t)3 * HDIM * DDIM * 2);  // 3 x D x H
    float*    Abuf  = (float*)alloc((size_t)NTOT * HDIM * 4);
    float*    Bbuf  = (float*)alloc((size_t)NTOT * HDIM * 4);
    _Float16* H1h   = (_Float16*)alloc((size_t)NTOT * HDIM * 2);
    float*    SP    = (float*)alloc((size_t)NTOT * DDIM * 4);
    int*   cnt   = (int*)alloc((size_t)NTOT * 4);
    int*   smass = (int*)alloc((size_t)NTOT * 4);
    int*   sfi   = (int*)alloc((size_t)NTOT * 4);
    int*   efi   = (int*)alloc((size_t)NTOT * 4);
    float* ninf  = (float*)alloc((size_t)NTOT * 4);

    int n;
    n = NTOT * DDIM;
    f32_to_f16_kernel<<<(n + 255) / 256, 256, 0, stream>>>(states, Sh, n);
    n = 3 * DDIM * HDIM;
    convert_transpose_kernel<<<(n + 255) / 256, 256, 0, stream>>>(W_out, WoutT, DDIM, HDIM);
    convert_transpose_kernel<<<(n + 255) / 256, 256, 0, stream>>>(W_in,  WinT,  DDIM, HDIM);
    convert_transpose_kernel<<<(n + 255) / 256, 256, 0, stream>>>(W1,    W1T,   DDIM, HDIM);
    convert_transpose_kernel<<<(n + 255) / 256, 256, 0, stream>>>(W2,    W2T,   HDIM, DDIM);

    dim3 blk(32, 4);
    dim3 gA(NTOT / 16, HDIM / 256);  // 288 x 2 blocks, each wave: 16x64 strip
    wmma_gemm_kernel<<<gA, blk, 0, stream>>>(Sh, WoutT, b_out, Abuf, nullptr, DDIM, HDIM, 1);
    wmma_gemm_kernel<<<gA, blk, 0, stream>>>(Sh, WinT,  b_in,  Bbuf, nullptr, DDIM, HDIM, 1);
    wmma_gemm_kernel<<<gA, blk, 0, stream>>>(Sh, W1T,   b1,    nullptr, H1h,  DDIM, HDIM, 1);
    dim3 gS(NTOT / 16, DDIM / 256);  // 288 x 1 blocks, K = 512
    wmma_gemm_kernel<<<gS, blk, 0, stream>>>(H1h, W2T,  b2,    SP,   nullptr, HDIM, DDIM, 0);

    neighbor_kernel<<<NTOT / 8, 256, 0, stream>>>(positions, roles, Abuf, Bbuf,
                                                  cnt, smass, sfi, efi, ninf);
    finalize_kernel<<<NTOT, 256, 0, stream>>>(states, energies, rgrid, roles, positions,
                                              SP, cnt, smass, sfi, efi, ninf, out);
}
